// RoPESDPAAttention_78683800863198
// MI455X (gfx1250) — matrix operand order repaired and checked by reading, not compiled, not run on hardware
//
#include <hip/hip_runtime.h>
#include <hip/hip_bf16.h>
#include <stdint.h>

// ---------------------------------------------------------------------------
// RoPE + SDPA attention for MI455X (gfx1250), bf16 WMMA pipeline.
//   B=8, L=1024, HIDDEN=512, NHEAD=8, HEAD_DIM=64
// All matmuls use v_wmma_f32_16x16x32_bf16 (f32 accumulate).
// ---------------------------------------------------------------------------

#define B_    8
#define L_    1024
#define H_    8
#define DH_   64
#define HID_  512

typedef __bf16 v16bf __attribute__((ext_vector_type(16)));
typedef float  v8f   __attribute__((ext_vector_type(8)));

union BFrag {
    v16bf v;
    uint4 u[2];
};

// A-operand (16x32 bf16, MxK): lanes 0-15 hold row M=lane, K k0..k0+15;
// lanes 16-31 hold row M=lane-16, K k0+16..k0+31.
__device__ __forceinline__ BFrag load_fragA(const __bf16* base, int row, int ld,
                                            int k0, int hf) {
    const __bf16* p = base + (size_t)row * ld + k0 + hf * 16;
    BFrag f;
    f.u[0] = *(const uint4*)(p);
    f.u[1] = *(const uint4*)(p + 8);
    return f;
}

// B-operand (32x16 bf16, KxN), loaded from row-major B^T storage ([N][K]):
// lane holds column N=lane&15; lanes 0-15 carry K=k0..k0+15, lanes 16-31 K=k0+16..k0+31.
__device__ __forceinline__ BFrag load_fragB(const __bf16* base, int rowN, int ld,
                                            int k0, int hf) {
    const __bf16* p = base + (size_t)rowN * ld + k0 + hf * 16;
    BFrag f;
    f.u[0] = *(const uint4*)(p);
    f.u[1] = *(const uint4*)(p + 8);
    return f;
}

__device__ __forceinline__ v8f wmma_bf16(v16bf a, v16bf b, v8f c) {
    return __builtin_amdgcn_wmma_f32_16x16x32_bf16(false, a, false, b,
                                                   (short)0, c, false, false);
}

// ---------------------------------------------------------------------------
// fp32 -> bf16 conversion (vectorized x4)
// ---------------------------------------------------------------------------
__global__ void cvt_f32_bf16(const float* __restrict__ in, __bf16* __restrict__ out,
                             int n4) {
    int i = blockIdx.x * blockDim.x + threadIdx.x;
    if (i < n4) {
        float4 v = ((const float4*)in)[i];
        union { uint2 u; __bf16 h[4]; } pk;
        pk.h[0] = (__bf16)v.x; pk.h[1] = (__bf16)v.y;
        pk.h[2] = (__bf16)v.z; pk.h[3] = (__bf16)v.w;
        ((uint2*)out)[i] = pk.u;
    }
}

// ---------------------------------------------------------------------------
// QKV projection + RoPE.
//   grid = (512 row-tiles, 3 types: 0=Q 1=K 2=V), block = 256 (8 waves = 8 heads)
//   Each wave: 16 rows x 64 cols (one head) strip, K-loop 512/32 = 16 WMMAs.
//   Q gets extra 1/sqrt(64) scale folded in. Q/K -> [B,H,L,Dh] bf16 (LDS repack),
//   V -> transposed [B,H,Dh,L] bf16 directly from accumulator (b128 stores).
// ---------------------------------------------------------------------------
__global__ __launch_bounds__(256) void qkv_rope(
    const __bf16* __restrict__ xb,
    const __bf16* __restrict__ Wqb, const __bf16* __restrict__ Wkb,
    const __bf16* __restrict__ Wvb,
    __bf16* __restrict__ Qb, __bf16* __restrict__ Kb, __bf16* __restrict__ Vt) {

    __shared__ float lds[8][16 * 68];   // per-wave 16x64 f32 tile, padded stride 68

    const int wave = threadIdx.x >> 5;
    const int lid  = threadIdx.x & 31;
    const int hf   = lid >> 4;          // lane half
    const int n    = lid & 15;          // column within tile
    const int rowTile = blockIdx.x;     // 0..511 over B*L/16
    const int type    = blockIdx.y;     // 0=Q 1=K 2=V
    const int h       = wave;           // head

    const __bf16* W = (type == 0) ? Wqb : ((type == 1) ? Wkb : Wvb);

    v8f acc[4] = {v8f{}, v8f{}, v8f{}, v8f{}};
    const int arow = rowTile * 16 + n;

    for (int k0 = 0; k0 < HID_; k0 += 32) {
        BFrag a = load_fragA(xb, arow, HID_, k0, hf);
#pragma unroll
        for (int t = 0; t < 4; t++) {
            BFrag b = load_fragB(W, h * DH_ + t * 16 + n, HID_, k0, hf);
            acc[t] = wmma_bf16(a.v, b.v, acc[t]);
        }
    }

    if (type < 2) {
        // RoPE in-register: dh = t*16 + n; i = dh & 31; partner tile = t^2.
        const float Cc = 0.2878231366242557f;              // ln(10000)/32
        const float invf0 = __expf(-(float)n * Cc);        // t even
        const float invf1 = __expf(-(float)(n + 16) * Cc); // t odd
        const float posBase = (float)(((rowTile & 63) * 16) + hf * 8);
        const float qs = (type == 0) ? 0.125f : 1.0f;      // fold 1/sqrt(64) into Q
#pragma unroll
        for (int r = 0; r < 8; r++) {
            float sn0, cs0, sn1, cs1;
            const float pos = posBase + (float)r;
            __sincosf(pos * invf0, &sn0, &cs0);
            __sincosf(pos * invf1, &sn1, &cs1);
            const float a0 = acc[0][r], a1 = acc[1][r];
            const float a2 = acc[2][r], a3 = acc[3][r];
            acc[0][r] = (a0 * cs0 - a2 * sn0) * qs;   // dh<32: rot = -x[dh+32]
            acc[1][r] = (a1 * cs1 - a3 * sn1) * qs;
            acc[2][r] = (a2 * cs0 + a0 * sn0) * qs;   // dh>=32: rot = +x[dh-32]
            acc[3][r] = (a3 * cs1 + a1 * sn1) * qs;
        }
        // LDS repack: C-layout (col per lane) -> row-major bf16 rows, b128 stores.
        float* Lp = lds[wave];
#pragma unroll
        for (int r = 0; r < 8; r++) {
            const int rowm = r + hf * 8;
#pragma unroll
            for (int t = 0; t < 4; t++)
                Lp[rowm * 68 + t * 16 + n] = acc[t][r];
        }
        __syncthreads();
        const int rr = n, cb = hf * 32;
        const int g  = rowTile * 16 + rr;
        const int bb = g >> 10, pos = g & 1023;
        __bf16* dst = ((type == 0) ? Qb : Kb)
                      + ((size_t)(bb * H_ + h) * L_ + pos) * DH_;
#pragma unroll
        for (int c = 0; c < 4; c++) {
            union { uint4 u; __bf16 hh[8]; } pk;
#pragma unroll
            for (int e = 0; e < 8; e++)
                pk.hh[e] = (__bf16)Lp[rr * 68 + cb + c * 8 + e];
            *(uint4*)(dst + cb + c * 8) = pk.u;
        }
    } else {
        // V: store transposed [B,H,Dh,L]; per lane the 8 accumulator rows are
        // 8 consecutive positions -> one b128 bf16 store per tile, no LDS.
        const int g = rowTile * 16;
        const int bb = g >> 10;
        const int posB = (g & 1023) + hf * 8;
#pragma unroll
        for (int t = 0; t < 4; t++) {
            const int dh = t * 16 + n;
            union { uint4 u; __bf16 hh[8]; } pk;
#pragma unroll
            for (int r = 0; r < 8; r++) pk.hh[r] = (__bf16)acc[t][r];
            *(uint4*)(Vt + ((size_t)(bb * H_ + h) * DH_ + dh) * L_ + posB) = pk.u;
        }
    }
}

// ---------------------------------------------------------------------------
// Flash attention: S^T = K*Q^T so each lane owns one query column ->
// softmax is in-lane + one shfl_xor(16); O^T = V^T * P^T.
//   grid = (16, B*H=64), block = 128 (4 waves, wave = query tile)
// ---------------------------------------------------------------------------
__global__ __launch_bounds__(128) void flash_attn(
    const __bf16* __restrict__ Qb, const __bf16* __restrict__ Kb,
    const __bf16* __restrict__ Vt, __bf16* __restrict__ Ab) {

    const int wave = threadIdx.x >> 5;
    const int lid  = threadIdx.x & 31;
    const int hf   = lid >> 4;
    const int n    = lid & 15;
    const int qTile = blockIdx.x * 4 + wave;  // 0..63
    const int bh    = blockIdx.y;             // b*8 + h
    const int qBase = qTile * 16;

    const __bf16* Qp = Qb + (size_t)bh * L_ * DH_;
    const __bf16* Kp = Kb + (size_t)bh * L_ * DH_;
    const __bf16* Vp = Vt + (size_t)bh * DH_ * L_;

    // Q as B-operand of S^T = K * Q^T (column = query), two dh chunks.
    const BFrag qf0 = load_fragB(Qp, qBase + n, DH_, 0, hf);
    const BFrag qf1 = load_fragB(Qp, qBase + n, DH_, 32, hf);

    v8f o[4] = {v8f{}, v8f{}, v8f{}, v8f{}};  // O^T, 4 dh tiles of 16
    float m = -1e30f, lsum = 0.f;

    for (int kb = 0; kb < L_; kb += 32) {
        // K as A-operand: 2 key sub-tiles x 2 dh chunks.
        const BFrag ka00 = load_fragA(Kp, kb + n,      DH_, 0,  hf);
        const BFrag ka01 = load_fragA(Kp, kb + n,      DH_, 32, hf);
        const BFrag ka10 = load_fragA(Kp, kb + 16 + n, DH_, 0,  hf);
        const BFrag ka11 = load_fragA(Kp, kb + 16 + n, DH_, 32, hf);
        v8f s0 = v8f{}, s1 = v8f{};
        s0 = wmma_bf16(ka00.v, qf0.v, s0);
        s0 = wmma_bf16(ka01.v, qf1.v, s0);
        s1 = wmma_bf16(ka10.v, qf0.v, s1);
        s1 = wmma_bf16(ka11.v, qf1.v, s1);

        // Online softmax (per-lane query column; keys striped across VGPRs).
        float bm = -1e30f;
#pragma unroll
        for (int i = 0; i < 8; i++) bm = fmaxf(bm, fmaxf(s0[i], s1[i]));
        bm = fmaxf(bm, __shfl_xor(bm, 16, 32));
        const float mn = fmaxf(m, bm);
        const float alpha = __expf(m - mn);
        m = mn;

        // exp + repack P^T into B-operand layout (half-swap via one shuffle pair).
        union { v16bf v; __bf16 hh[16]; } pf;
        float ls = 0.f;
#pragma unroll
        for (int i = 0; i < 8; i++) {
            const float e0 = __expf(s0[i] - mn);
            const float e1 = __expf(s1[i] - mn);
            ls += e0 + e1;
            const float x0 = __shfl_xor(e0, 16, 32);
            const float x1 = __shfl_xor(e1, 16, 32);
            pf.hh[i]     = (__bf16)(hf ? x1 : e0);  // K = k0..k0+7   of 32-block
            pf.hh[8 + i] = (__bf16)(hf ? e1 : x0);  // K = k0+8..k0+15
        }
        lsum = lsum * alpha + ls;
#pragma unroll
        for (int t = 0; t < 4; t++)
#pragma unroll
            for (int i = 0; i < 8; i++) o[t][i] *= alpha;

        // O^T += V^T(dh x keys) * P^T(keys x q); V^T rows contiguous in Vt.
#pragma unroll
        for (int t = 0; t < 4; t++) {
            const BFrag vf = load_fragA(Vp, t * 16 + n, L_, kb, hf);
            o[t] = wmma_bf16(vf.v, pf.v, o[t]);
        }
    }

    lsum += __shfl_xor(lsum, 16, 32);   // merge the two key half-sums
    const float rinv = 1.f / lsum;

    // O^T C-layout store: per lane 8 consecutive dh -> b128 bf16 stores.
    const int b = bh >> 3, h = bh & 7;
    __bf16* dst = Ab + ((size_t)b * L_ + qBase + n) * HID_ + h * DH_;
#pragma unroll
    for (int t = 0; t < 4; t++) {
        union { uint4 u; __bf16 hh[8]; } pk;
#pragma unroll
        for (int i = 0; i < 8; i++) pk.hh[i] = (__bf16)(o[t][i] * rinv);
        *(uint4*)(dst + t * 16 + hf * 8) = pk.u;
    }
}

// ---------------------------------------------------------------------------
// Output projection: out = Ab @ Wo^T (f32 out).
//   grid = 512 row-tiles, block = 256 (8 waves = 8 column strips of 64)
// ---------------------------------------------------------------------------
__global__ __launch_bounds__(256) void out_proj(
    const __bf16* __restrict__ Ab, const __bf16* __restrict__ Wob,
    float* __restrict__ out) {

    __shared__ float lds[8][16 * 68];

    const int wave = threadIdx.x >> 5;
    const int lid  = threadIdx.x & 31;
    const int hf   = lid >> 4;
    const int n    = lid & 15;
    const int rowTile = blockIdx.x;
    const int strip   = wave;

    v8f acc[4] = {v8f{}, v8f{}, v8f{}, v8f{}};
    const int arow = rowTile * 16 + n;
    for (int k0 = 0; k0 < HID_; k0 += 32) {
        BFrag a = load_fragA(Ab, arow, HID_, k0, hf);
#pragma unroll
        for (int t = 0; t < 4; t++) {
            BFrag b = load_fragB(Wob, strip * 64 + t * 16 + n, HID_, k0, hf);
            acc[t] = wmma_bf16(a.v, b.v, acc[t]);
        }
    }
    float* Lp = lds[wave];
#pragma unroll
    for (int r = 0; r < 8; r++) {
        const int rowm = r + hf * 8;
#pragma unroll
        for (int t = 0; t < 4; t++)
            Lp[rowm * 68 + t * 16 + n] = acc[t][r];
    }
    __syncthreads();
    const int rr = n, cb = hf * 32;
    float* dst = out + (size_t)(rowTile * 16 + rr) * HID_ + strip * 64;
#pragma unroll
    for (int c = 0; c < 8; c++)
        *(uint4*)(dst + cb + c * 4) = *(const uint4*)&Lp[rr * 68 + cb + c * 4];
}

// ---------------------------------------------------------------------------
// Host launcher
// ---------------------------------------------------------------------------
extern "C" void kernel_launch(void* const* d_in, const int* in_sizes, int n_in,
                              void* d_out, int out_size, void* d_ws, size_t ws_size,
                              hipStream_t stream) {
    const float* x  = (const float*)d_in[0];
    const float* Wq = (const float*)d_in[1];
    const float* Wk = (const float*)d_in[2];
    const float* Wv = (const float*)d_in[3];
    const float* Wo = (const float*)d_in[4];
    float* out = (float*)d_out;

    const size_t NX = (size_t)B_ * L_ * HID_;   // 4,194,304
    const size_t NW = (size_t)HID_ * HID_;      //   262,144

    char* w = (char*)d_ws;
    auto carve = [&](size_t elems) {
        __bf16* p = (__bf16*)w;
        w += ((elems * sizeof(__bf16)) + 255) & ~(size_t)255;
        return p;
    };
    __bf16* xb  = carve(NX);
    __bf16* Wqb = carve(NW);
    __bf16* Wkb = carve(NW);
    __bf16* Wvb = carve(NW);
    __bf16* Wob = carve(NW);
    __bf16* Qb  = carve(NX);   // [B,H,L,Dh]
    __bf16* Kb  = carve(NX);   // [B,H,L,Dh]
    __bf16* Vt  = carve(NX);   // [B,H,Dh,L]  (transposed V)
    __bf16* Ab  = carve(NX);   // [B,L,H*Dh]  attention output, bf16

    // 1) fp32 -> bf16 conversions
    {
        int n4 = (int)(NX / 4);
        cvt_f32_bf16<<<(n4 + 255) / 256, 256, 0, stream>>>(x, xb, n4);
        int w4 = (int)(NW / 4);
        cvt_f32_bf16<<<(w4 + 255) / 256, 256, 0, stream>>>(Wq, Wqb, w4);
        cvt_f32_bf16<<<(w4 + 255) / 256, 256, 0, stream>>>(Wk, Wkb, w4);
        cvt_f32_bf16<<<(w4 + 255) / 256, 256, 0, stream>>>(Wv, Wvb, w4);
        cvt_f32_bf16<<<(w4 + 255) / 256, 256, 0, stream>>>(Wo, Wob, w4);
    }

    // 2) QKV projection + RoPE (+ V transpose)
    qkv_rope<<<dim3(512, 3), 256, 0, stream>>>(xb, Wqb, Wkb, Wvb, Qb, Kb, Vt);

    // 3) Flash attention
    flash_attn<<<dim3(16, 64), 128, 0, stream>>>(Qb, Kb, Vt, Ab);

    // 4) Output projection
    out_proj<<<dim3(512, 1), 256, 0, stream>>>(Ab, Wob, out);
}
